// DirectionPolicyHead_22789096473205
// MI455X (gfx1250) — matrix operand order repaired and checked by reading, not compiled, not run on hardware
//
#include <hip/hip_runtime.h>

typedef unsigned short u16;
typedef __attribute__((ext_vector_type(16))) __bf16 v16bf;
typedef __attribute__((ext_vector_type(8)))  __bf16 bf16x8;
typedef __attribute__((ext_vector_type(8)))  float  v8f;
typedef __attribute__((ext_vector_type(8)))  u16    u16x8;

// ---------- helpers ----------
__device__ __forceinline__ u16 f2bf(float f) {            // f32 -> bf16 RNE
  unsigned u = __float_as_uint(f);
  u += 0x7FFFu + ((u >> 16) & 1u);
  return (u16)(u >> 16);
}

// build a v16bf WMMA fragment from two 16-byte LDS chunks
__device__ __forceinline__ v16bf ldv16(const u16* p0, const u16* p1) {
  bf16x8 a = __builtin_bit_cast(bf16x8, *(const u16x8*)p0);
  bf16x8 b = __builtin_bit_cast(bf16x8, *(const u16x8*)p1);
  v16bf r;
#pragma unroll
  for (int i = 0; i < 8; ++i) { r[i] = a[i]; r[i + 8] = b[i]; }
  return r;
}

__device__ __forceinline__ v8f wmma_bf16(v16bf a, v16bf b, v8f c) {
  // D = A(16x32 bf16) * B(32x16 bf16) + C(16x16 f32)
  return __builtin_amdgcn_wmma_f32_16x16x32_bf16(false, a, false, b, (short)0, c,
                                                 false, false);
}

// order-preserving float<->u32 encoding so max() works as unsigned atomicMax
__device__ __forceinline__ unsigned encf(float f) {
  unsigned u = __float_as_uint(f);
  return (u >> 31) ? ~u : (u | 0x80000000u);
}
__device__ __forceinline__ float decf(unsigned e) {
  unsigned u = (e >> 31) ? (e ^ 0x80000000u) : ~e;
  return __uint_as_float(u);
}

#define NEGV (-10000000000.0f)
#define POLICY_SIZE 2187
#define MOVE_POLICY 1620   // 20 directions * 81, scatter-max region
#define DROP_START 1620

// ---------- init: bf16-transposed weights + padded drop queries ----------
__global__ void k_prep(const float* __restrict__ We, const float* __restrict__ Wq,
                       const float* __restrict__ Wk, const float* __restrict__ dq,
                       u16* __restrict__ WtE, u16* __restrict__ WtQ,
                       u16* __restrict__ WtK, u16* __restrict__ DQ) {
  int idx = blockIdx.x * 256 + threadIdx.x;
  if (idx < 3 * 65536) {
    int w = idx >> 16, rem = idx & 65535;
    int n = rem >> 8, k = rem & 255;
    const float* W = (w == 0) ? We : ((w == 1) ? Wq : Wk);
    u16* Wt = (w == 0) ? WtE : ((w == 1) ? WtQ : WtK);
    Wt[n * 256 + k] = f2bf(W[k * 256 + n]);       // transpose: [N][K]
  } else if (idx < 3 * 65536 + 16 * 256) {
    int rem = idx - 3 * 65536;
    int q = rem >> 8, k = rem & 255;
    DQ[rem] = (q < 7) ? f2bf(dq[q * 256 + k]) : (u16)0;
  }
}

__global__ void k_map_init(unsigned* __restrict__ MAP) {
  int i = blockIdx.x * 256 + threadIdx.x;
  if (i < 6561) MAP[i] = 0xFFFFFFFFu;
}

__global__ void k_map_fill(const int* __restrict__ vi, const int* __restrict__ pi,
                           const int* __restrict__ qi, const int* __restrict__ qpi,
                           int nv, int np, u16* __restrict__ MAPh) {
  int i = blockIdx.x * 256 + threadIdx.x;
  if (i < nv) {
    MAPh[(size_t)vi[i] * 2 + 0] = (u16)pi[i];
  } else if (i < nv + np) {
    int j = i - nv;
    MAPh[(size_t)qi[j] * 2 + 1] = (u16)qpi[j];
  }
}

// ---------- stage A: pe = mish(x@We+bE); Q = pe@Wq+bq; K = pe@Wk+bk ----------
#define MT 128

__global__ __launch_bounds__(256) void dph_stageA(
    const float* __restrict__ x,
    const float* __restrict__ bE, const float* __restrict__ bQ,
    const float* __restrict__ bK,
    const u16* __restrict__ WtE, const u16* __restrict__ WtQ,
    const u16* __restrict__ WtK,
    u16* __restrict__ Qo, u16* __restrict__ Ko) {
  __shared__ __align__(16) u16 ldsA[MT * 256];   // 64 KB: x(bf16) -> pe(bf16)
  __shared__ __align__(16) u16 ldsW[256 * 128];  // 64 KB: weight half / out staging

  const int tid  = threadIdx.x;
  const int lane = tid & 31;
  const int wave = tid >> 5;
  const int l16  = lane & 15;
  const int khalf = lane >> 4;
  const size_t m0 = (size_t)blockIdx.x * MT;

  // load x tile, convert to bf16 in LDS (fully coalesced: tile is contiguous)
  {
    const float4* gx = (const float4*)(x + m0 * 256);
    uint2* la = (uint2*)ldsA;
    for (int v = tid; v < MT * 256 / 4; v += 256) {
      float4 f = gx[v];
      uint2 p;
      p.x = (unsigned)f2bf(f.x) | ((unsigned)f2bf(f.y) << 16);
      p.y = (unsigned)f2bf(f.z) | ((unsigned)f2bf(f.w) << 16);
      la[v] = p;
    }
  }
  // warm L2 for the later weight phases (global_prefetch_b8)
  {
    const u16* pq = WtQ + (size_t)tid * 256;
    const u16* pk = WtK + (size_t)tid * 256;
#pragma unroll
    for (int i = 0; i < 4; ++i) {
      __builtin_prefetch(pq + i * 64, 0, 0);
      __builtin_prefetch(pk + i * 64, 0, 0);
    }
  }

  auto loadWhalf = [&](const u16* Wt, int h) {   // [256 n][128 k] half into LDS
    const uint4* src = (const uint4*)(Wt + (size_t)tid * 256 + h * 128);
    uint4* dst = (uint4*)(ldsW + tid * 128);
#pragma unroll
    for (int i = 0; i < 16; ++i) dst[i] = src[i];
  };

  v8f acc[16];                                   // 16 rows x 256 cols per wave
  auto zeroAcc = [&]() {
#pragma unroll
    for (int nt = 0; nt < 16; ++nt)
#pragma unroll
      for (int r = 0; r < 8; ++r) acc[nt][r] = 0.f;
  };

  // ks kept as a real loop (unroll 1): only one k-step's LDS addresses live at
  // a time -> single base VGPR + DS offset immediates for the 16 B fragments,
  // leaving the scheduler registers to keep several ds_load_b128 in flight.
  auto gemmHalf = [&](int half) {
    const u16* Arow = ldsA + (wave * 16 + l16) * 256 + half * 128 + khalf * 8;
    const u16* wbase = ldsW + l16 * 128 + khalf * 8;
#pragma unroll 1
    for (int ks = 0; ks < 4; ++ks) {
      v16bf a = ldv16(Arow + ks * 32, Arow + ks * 32 + 16);
      const u16* wp = wbase + ks * 32;
#pragma unroll
      for (int nt = 0; nt < 16; ++nt) {
        v16bf b = ldv16(wp + nt * 2048, wp + nt * 2048 + 16);
        acc[nt] = wmma_bf16(a, b, acc[nt]);
      }
    }
  };

  // ===== pe = mish(x @ We + bE) =====
  zeroAcc();
  loadWhalf(WtE, 0);
  __syncthreads();
  gemmHalf(0);
  __syncthreads();
  loadWhalf(WtE, 1);
  __syncthreads();
  gemmHalf(1);

#pragma unroll
  for (int nt = 0; nt < 16; ++nt) {
    const int col = nt * 16 + l16;
    const float bv = bE[col];
#pragma unroll
    for (int r = 0; r < 8; ++r) {
      float v = acc[nt][r] + bv;
      float sp = (v > 15.f) ? v : log1pf(__expf(v));  // softplus
      float pe = v * tanhf(sp);                       // mish
      const int row = wave * 16 + r + 8 * khalf;      // C layout: M = r + 8*khalf
      ldsA[row * 256 + col] = f2bf(pe);               // wave-private rows
    }
  }

  // ===== Q = pe @ Wq + bq =====
  __syncthreads();
  loadWhalf(WtQ, 0);
  __syncthreads();
  zeroAcc();
  gemmHalf(0);
  __syncthreads();
  loadWhalf(WtQ, 1);
  __syncthreads();
  gemmHalf(1);
  __syncthreads();
#pragma unroll
  for (int nt = 0; nt < 16; ++nt) {
    const int col = nt * 16 + l16;
    const float bv = bQ[col];
#pragma unroll
    for (int r = 0; r < 8; ++r) {
      const int row = wave * 16 + r + 8 * khalf;
      ldsW[row * 256 + col] = f2bf(acc[nt][r] + bv);  // stage Q tile in LDS
    }
  }
  __syncthreads();
  {
    const uint4* s = (const uint4*)ldsW;
    uint4* d = (uint4*)(Qo + m0 * 256);
    for (int v = tid; v < MT * 256 / 8; v += 256) d[v] = s[v];  // coalesced out
  }
  __syncthreads();

  // ===== K = pe @ Wk + bk =====
  loadWhalf(WtK, 0);
  __syncthreads();
  zeroAcc();
  gemmHalf(0);
  __syncthreads();
  loadWhalf(WtK, 1);
  __syncthreads();
  gemmHalf(1);
  __syncthreads();
#pragma unroll
  for (int nt = 0; nt < 16; ++nt) {
    const int col = nt * 16 + l16;
    const float bv = bK[col];
#pragma unroll
    for (int r = 0; r < 8; ++r) {
      const int row = wave * 16 + r + 8 * khalf;
      ldsW[row * 256 + col] = f2bf(acc[nt][r] + bv);
    }
  }
  __syncthreads();
  {
    const uint4* s = (const uint4*)ldsW;
    uint4* d = (uint4*)(Ko + m0 * 256);
    for (int v = tid; v < MT * 256 / 8; v += 256) d[v] = s[v];
  }
}

// ---------- stage B: board = Q K^T * scale; scatter-max; drop logits ----------
__global__ __launch_bounds__(256) void dph_stageB(
    const u16* __restrict__ Q, const u16* __restrict__ K,
    const u16* __restrict__ DQ, const unsigned* __restrict__ MAP,
    float* __restrict__ out) {
  __shared__ __align__(16) u16 ldsQ[96 * 256];   // 48 KB (rows 81..95 zero)
  __shared__ __align__(16) u16 ldsK[96 * 256];   // 48 KB
  __shared__ __align__(16) u16 ldsDQ[16 * 256];  // 8 KB
  __shared__ unsigned ldsM[6561];                // ft -> (policy | promo<<16)
  __shared__ unsigned ldsP[MOVE_POLICY];         // encoded policy maxima

  const int b = blockIdx.x;
  const int tid = threadIdx.x;
  const int lane = tid & 31;
  const int wave = tid >> 5;
  const int l16 = lane & 15;
  const int khalf = lane >> 4;
  const float scale = 0.0625f;                   // 1/sqrt(256)

  // cooperative loads (all contiguous, uint4-wide)
  {
    const uint4* gq = (const uint4*)(Q + (size_t)b * 81 * 256);
    const uint4* gk = (const uint4*)(K + (size_t)b * 81 * 256);
    uint4* lq = (uint4*)ldsQ;
    uint4* lk = (uint4*)ldsK;
    uint4 z4; z4.x = z4.y = z4.z = z4.w = 0u;
    for (int v = tid; v < 2592; v += 256) { lq[v] = gq[v]; lk[v] = gk[v]; }
    for (int v = tid; v < 480; v += 256) { lq[2592 + v] = z4; lk[2592 + v] = z4; }
    const uint4* gd = (const uint4*)DQ;
    uint4* ld = (uint4*)ldsDQ;
    for (int v = tid; v < 512; v += 256) ld[v] = gd[v];
  }
  for (int v = tid; v < 6561; v += 256) ldsM[v] = MAP[v];
  {
    const unsigned negE = encf(NEGV);
    for (int v = tid; v < MOVE_POLICY; v += 256) ldsP[v] = negE;
  }
  __syncthreads();

  // 36 board tiles (6x6 of 16x16 over 96x96) + 6 drop tiles, split over 8 waves
  for (int tIdx = wave; tIdx < 42; tIdx += 8) {
    int i, j;
    const u16* Arow;
    if (tIdx < 36) {
      i = tIdx / 6; j = tIdx % 6;
      Arow = ldsQ + (i * 16 + l16) * 256 + khalf * 8;
    } else {
      i = -1; j = tIdx - 36;
      Arow = ldsDQ + l16 * 256 + khalf * 8;
    }
    const u16* Brow = ldsK + (j * 16 + l16) * 256 + khalf * 8;

    v8f acc;
#pragma unroll
    for (int r = 0; r < 8; ++r) acc[r] = 0.f;

#pragma unroll
    for (int ks = 0; ks < 8; ++ks) {
      const int k = ks * 32;
      v16bf a = ldv16(Arow + k, Arow + k + 16);
      v16bf bb = ldv16(Brow + k, Brow + k + 16);
      acc = wmma_bf16(a, bb, acc);
    }

    if (tIdx < 36) {
#pragma unroll
      for (int r = 0; r < 8; ++r) {
        const int s = i * 16 + r + 8 * khalf;
        const int t = j * 16 + l16;
        if (s < 81 && t < 81) {
          const unsigned e = encf(acc[r] * scale);
          const unsigned m = ldsM[s * 81 + t];
          const unsigned p0 = m & 0xFFFFu;
          const unsigned p1 = m >> 16;
          if (p0 != 0xFFFFu) atomicMax(&ldsP[p0], e);
          if (p1 != 0xFFFFu) atomicMax(&ldsP[p1], e);
        }
      }
    } else {
#pragma unroll
      for (int r = 0; r < 8; ++r) {
        const int q = r + 8 * khalf;
        const int t = j * 16 + l16;
        if (q < 7 && t < 81)
          out[(size_t)b * POLICY_SIZE + DROP_START + q * 81 + t] = acc[r] * scale;
      }
    }
  }
  __syncthreads();

  for (int v = tid; v < MOVE_POLICY; v += 256)
    out[(size_t)b * POLICY_SIZE + v] = decf(ldsP[v]);
}

// ---------- launcher ----------
extern "C" void kernel_launch(void* const* d_in, const int* in_sizes, int n_in,
                              void* d_out, int out_size, void* d_ws, size_t ws_size,
                              hipStream_t stream) {
  (void)n_in; (void)out_size; (void)ws_size;
  const float* x  = (const float*)d_in[0];
  const float* We = (const float*)d_in[1];
  const float* bE = (const float*)d_in[2];
  const float* Wq = (const float*)d_in[3];
  const float* bq = (const float*)d_in[4];
  const float* Wk = (const float*)d_in[5];
  const float* bk = (const float*)d_in[6];
  const float* dq = (const float*)d_in[7];
  const int* vi   = (const int*)d_in[8];
  const int* pi   = (const int*)d_in[9];
  const int* qi   = (const int*)d_in[10];
  const int* qpi  = (const int*)d_in[11];
  const int nv = in_sizes[8];
  const int np = in_sizes[10];

  const int B = in_sizes[0] / (81 * 256);   // 4096
  const int Mtotal = B * 81;                // 331776, divisible by 128

  char* ws = (char*)d_ws;
  u16* WtE = (u16*)(ws + 0);                       // 128 KB
  u16* WtQ = (u16*)(ws + 131072);                  // 128 KB
  u16* WtK = (u16*)(ws + 262144);                  // 128 KB
  u16* DQb = (u16*)(ws + 393216);                  // 8 KB (16x256 bf16)
  unsigned* MAP = (unsigned*)(ws + 401408);        // 26.2 KB
  u16* Qb = (u16*)(ws + 430080);                   // Mtotal*256 bf16
  u16* Kb = (u16*)(ws + 430080 + (size_t)Mtotal * 256 * 2);
  float* out = (float*)d_out;

  k_prep<<<(3 * 65536 + 16 * 256 + 255) / 256, 256, 0, stream>>>(
      We, Wq, Wk, dq, WtE, WtQ, WtK, DQb);
  k_map_init<<<(6561 + 255) / 256, 256, 0, stream>>>(MAP);
  k_map_fill<<<(nv + np + 255) / 256, 256, 0, stream>>>(vi, pi, qi, qpi, nv, np,
                                                        (u16*)MAP);
  dph_stageA<<<Mtotal / MT, 256, 0, stream>>>(x, bE, bq, bk, WtE, WtQ, WtK, Qb, Kb);
  dph_stageB<<<B, 256, 0, stream>>>(Qb, Kb, DQb, MAP, out);
}
